// HeteroHGT_3951369912437
// MI455X (gfx1250) — compile-verified, matrix operand order checked by reading, not run on hardware
//
#include <hip/hip_runtime.h>

// ---------------- problem constants (match reference) ----------------
#define LAYERS 2
#define TYPES  2
#define CH     128      // channels
#define HEADS  4
#define HD     32       // head dim
#define NNODE  50000
#define NEDGE  200000
#define MT     (NNODE / 16)   // 3125 row tiles, exact

typedef __attribute__((ext_vector_type(16))) __bf16 bf16x16;
typedef __attribute__((ext_vector_type(8)))  float  f32x8;

// ---------------- fragment packing helpers (CDNA5 16x16x32 bf16) ----------------
// A 16x32: lane m = lane&15; element i -> K = i + 8*((i>=8)+(lane>=16))
// Per lane: two 32B runs, cols [kb, kb+8) and [kb+16, kb+24), kb = 8*(lane>=16).

__device__ __forceinline__ bf16x16 pack_frag(float4 a0, float4 a1, float4 b0, float4 b1) {
  bf16x16 f;
  f[0]  = (__bf16)a0.x; f[1]  = (__bf16)a0.y; f[2]  = (__bf16)a0.z; f[3]  = (__bf16)a0.w;
  f[4]  = (__bf16)a1.x; f[5]  = (__bf16)a1.y; f[6]  = (__bf16)a1.z; f[7]  = (__bf16)a1.w;
  f[8]  = (__bf16)b0.x; f[9]  = (__bf16)b0.y; f[10] = (__bf16)b0.z; f[11] = (__bf16)b0.w;
  f[12] = (__bf16)b1.x; f[13] = (__bf16)b1.y; f[14] = (__bf16)b1.z; f[15] = (__bf16)b1.w;
  return f;
}

__device__ __forceinline__ bf16x16 load_frag_a(const float* __restrict__ A, int row0,
                                               int lda, int k0, int lane) {
  const float* p = A + (size_t)(row0 + (lane & 15)) * lda + k0 + ((lane >> 4) << 3);
  const float4* p4 = (const float4*)p;            // 16B aligned
  return pack_frag(p4[0], p4[1], p4[4], p4[5]);   // +0, +4, +16, +20
}

__device__ __forceinline__ float4 gelu4(float4 v) {
  float4 r;
  r.x = 0.5f * v.x * (1.0f + erff(v.x * 0.70710678118f));
  r.y = 0.5f * v.y * (1.0f + erff(v.y * 0.70710678118f));
  r.z = 0.5f * v.z * (1.0f + erff(v.z * 0.70710678118f));
  r.w = 0.5f * v.w * (1.0f + erff(v.w * 0.70710678118f));
  return r;
}

__device__ __forceinline__ bf16x16 load_frag_a_gelu(const float* __restrict__ A, int row0,
                                                    int lda, int k0, int lane) {
  const float* p = A + (size_t)(row0 + (lane & 15)) * lda + k0 + ((lane >> 4) << 3);
  const float4* p4 = (const float4*)p;
  return pack_frag(gelu4(p4[0]), gelu4(p4[1]), gelu4(p4[4]), gelu4(p4[5]));
}

// Packed B fragments: layout [ntile][kstep][lane][16 bf16] (32 B per lane, aligned).
__device__ __forceinline__ bf16x16 load_frag_b_packed(const __bf16* __restrict__ P,
                                                      int K, int n0, int k0, int lane) {
  return *(const bf16x16*)(P + ((size_t)((n0 >> 4) * (K >> 5) + (k0 >> 5)) * 32 + lane) * 16);
}

// C/D 16x16 f32: element r -> row r (+8 for upper half-wave), col = lane&15
__device__ __forceinline__ void store_frag(float* __restrict__ O, int row0, int ldo,
                                           int n0, int lane, f32x8 acc) {
  int n = lane & 15;
  int mb = (lane >> 4) << 3;
#pragma unroll
  for (int r = 0; r < 8; ++r)
    O[(size_t)(row0 + mb + r) * ldo + n0 + n] = acc[r];
}

__device__ __forceinline__ f32x8 wmma_bf16(bf16x16 a, bf16x16 b, f32x8 c) {
  return __builtin_amdgcn_wmma_f32_16x16x32_bf16(false, a, false, b, (short)0, c,
                                                 false, false);
}

__device__ __forceinline__ void atomicMaxFloat(float* addr, float val) {
  if (val >= 0.0f) atomicMax((int*)addr, __float_as_int(val));
  else             atomicMin((unsigned int*)addr, __float_as_uint(val));
}

// ---------------- kernels ----------------

__global__ __launch_bounds__(256)
void fill_kernel(float* __restrict__ p, float v, int n) {
  int i = blockIdx.x * 256 + threadIdx.x;
  if (i < n) p[i] = v;
}

// Convert a stack of row-major [K x Nn] fp32 matrices into bf16 WMMA B-fragment
// order: per matrix, [ntile][kstep][lane][16].  One thread per (ntile,kstep,lane).
__global__ __launch_bounds__(256)
void pack_b_kernel(const float* __restrict__ B, __bf16* __restrict__ P, int K, int Nn) {
  const int m = blockIdx.y;                       // matrix index in stack
  const int t = blockIdx.x * 256 + threadIdx.x;
  const int ntiles = Nn >> 4, ksteps = K >> 5;
  if (t >= ntiles * ksteps * 32) return;
  const int lane = t & 31;
  const int ks = (t >> 5) % ksteps;
  const int nt = (t >> 5) / ksteps;
  const float* Bm = B + (size_t)m * K * Nn;
  __bf16* q = P + (size_t)m * K * Nn + ((size_t)(nt * ksteps + ks) * 32 + lane) * 16;
  const float* p = Bm + (size_t)((ks << 5) + ((lane >> 4) << 3)) * Nn + (nt << 4) + (lane & 15);
#pragma unroll
  for (int i = 0; i < 8; ++i) {
    q[i]     = (__bf16)p[(size_t)i * Nn];
    q[i + 8] = (__bf16)p[(size_t)(i + 16) * Nn];
  }
}

// Out[M,N] = A[M,K] @ B[K,N] + bias, B pre-packed bf16.
// 4-wave workgroup: one 16-row A tile shared via async-to-LDS staging; each wave
// computes one 16x16 N tile.  K multiple of 32.  grid = (M/16, N/64).
__global__ __launch_bounds__(128)
void gemm_bias_kernel(const float* __restrict__ A, const __bf16* __restrict__ Bp,
                      const float* __restrict__ bias, float* __restrict__ Out,
                      int K, int lda, int ldo) {
  __shared__ float sA[16 * 32];                 // one 16x32 fp32 A tile (2 KB)
  const int t = threadIdx.x;                    // 0..127
  const int lane = t & 31;
  const int wid = t >> 5;                       // 0..3
  const int row0 = blockIdx.x * 16;
  const int n0 = (blockIdx.y * 4 + wid) * 16;

  // async staging: thread t moves one 16B chunk: row = t/8, cols [4*(t&7), +4)
  const int arow = t >> 3;
  const int acol = (t & 7) << 2;
  const unsigned lds_off = (unsigned)(size_t)&sA[arow * 32 + acol];

  f32x8 acc = {};
  for (int k0 = 0; k0 < K; k0 += 32) {
    const float* gsrc = A + (size_t)(row0 + arow) * lda + k0 + acol;
    // CDNA5 async copy global -> LDS (ASYNCcnt-tracked), one B128 per lane.
    asm volatile("global_load_async_to_lds_b128 %0, %1, off"
                 :: "v"(lds_off), "v"(gsrc) : "memory");
    if (k0 + 32 < K)   // prefetch next packed B k-slice
      __builtin_prefetch(Bp + ((size_t)((n0 >> 4) * (K >> 5) + ((k0 + 32) >> 5)) * 32 + lane) * 16, 0, 1);
    asm volatile("s_wait_asynccnt 0" ::: "memory");
    __syncthreads();

    const float* lp = &sA[(lane & 15) * 32 + ((lane >> 4) << 3)];
    const float4* lp4 = (const float4*)lp;
    bf16x16 a = pack_frag(lp4[0], lp4[1], lp4[4], lp4[5]);
    bf16x16 b = load_frag_b_packed(Bp, K, n0, k0, lane);
    acc = wmma_bf16(a, b, acc);
    __syncthreads();                            // before next k-step overwrites sA
  }

  int n = lane & 15;
  int mb = (lane >> 4) << 3;
  float bv = bias[n0 + n];
#pragma unroll
  for (int r = 0; r < 8; ++r)
    Out[(size_t)(row0 + mb + r) * ldo + n0 + n] = acc[r] + bv;
}

// Per-head 32x32 relation transform, in-place on K (which=0) or V (which=1)
// section of the source type's KQV buffer.  blockIdx: (mtile, head, e*2+which).
__global__ __launch_bounds__(32)
void rel_kernel(float* __restrict__ kqv0, float* __restrict__ kqv1,
                const __bf16* __restrict__ PA, const __bf16* __restrict__ PM, int l) {
  const int lane = threadIdx.x & 31;
  const int row0 = blockIdx.x * 16;
  const int h = blockIdx.y;
  const int e = blockIdx.z >> 1;       // relation; src type == e
  const int which = blockIdx.z & 1;    // 0 = K section, 1 = V section
  float* kqv = e ? kqv1 : kqv0;
  const int colbase = (which ? 2 * CH : 0) + h * HD;
  const __bf16* R = (which ? PM : PA) + (size_t)((l * 2 + e) * HEADS + h) * HD * HD;

  bf16x16 a  = load_frag_a(kqv, row0, 3 * CH, colbase, lane);
  bf16x16 b0 = load_frag_b_packed(R, HD, 0, 0, lane);
  bf16x16 b1 = load_frag_b_packed(R, HD, 16, 0, lane);
  f32x8 z = {};
  f32x8 d0 = wmma_bf16(a, b0, z);
  f32x8 d1 = wmma_bf16(a, b1, z);
  store_frag(kqv, row0, 3 * CH, colbase, lane, d0);
  store_frag(kqv, row0, 3 * CH, colbase + 16, lane, d1);
}

// Attention logits: wave per edge; lane covers 4 channels; 8-lane head reduce.
__global__ __launch_bounds__(256)
void alpha_kernel(const float* __restrict__ kqv0, const float* __restrict__ kqv1,
                  const int* __restrict__ e0, const int* __restrict__ e1,
                  const float* __restrict__ p_rel,
                  float* __restrict__ al0, float* __restrict__ al1,
                  float* __restrict__ am0, float* __restrict__ am1, int l) {
  const int e = blockIdx.y;
  const int lane = threadIdx.x & 31;
  const int edge = blockIdx.x * 8 + (threadIdx.x >> 5);
  const int* ed = e ? e1 : e0;
  const float* kqvS = e ? kqv1 : kqv0;  // source type == e
  const float* kqvD = e ? kqv0 : kqv1;  // dst type == 1-e
  int src = ed[edge], dst = ed[NEDGE + edge];
  int c = lane * 4;
  const float4 q4 = *(const float4*)(kqvD + (size_t)dst * 3 * CH + CH + c);
  const float4 k4 = *(const float4*)(kqvS + (size_t)src * 3 * CH + c);
  float s = q4.x * k4.x + q4.y * k4.y + q4.z * k4.z + q4.w * k4.w;
  s += __shfl_xor(s, 4, 32);
  s += __shfl_xor(s, 2, 32);
  s += __shfl_xor(s, 1, 32);
  if ((lane & 7) == 0) {
    int h = lane >> 3;
    float a = s * p_rel[(l * 2 + e) * HEADS + h] * 0.17677669529663687f; // 1/sqrt(32)
    (e ? al1 : al0)[(size_t)edge * HEADS + h] = a;
    atomicMaxFloat((e ? am0 : am1) + (size_t)dst * HEADS + h, a);
  }
}

// exp(a - segmax) ; accumulate denominators.
__global__ __launch_bounds__(256)
void exp_kernel(const int* __restrict__ e0, const int* __restrict__ e1,
                float* __restrict__ al0, float* __restrict__ al1,
                const float* __restrict__ am0, const float* __restrict__ am1,
                float* __restrict__ dn0, float* __restrict__ dn1) {
  const int e = blockIdx.y;
  int tid = blockIdx.x * 256 + threadIdx.x;
  int edge = tid >> 2, h = tid & 3;
  const int* ed = e ? e1 : e0;
  int dst = ed[NEDGE + edge];
  float* al = e ? al1 : al0;
  const float* am = e ? am0 : am1;
  float* dn = e ? dn0 : dn1;
  float ex = __expf(al[tid] - am[(size_t)dst * HEADS + h]);
  al[tid] = ex;
  atomicAdd(dn + (size_t)dst * HEADS + h, ex);
}

// Weighted message scatter-add: wave per edge, 4 channels/lane, L2 f32 atomics.
__global__ __launch_bounds__(256)
void agg_kernel(const float* __restrict__ kqv0, const float* __restrict__ kqv1,
                const int* __restrict__ e0, const int* __restrict__ e1,
                const float* __restrict__ al0, const float* __restrict__ al1,
                const float* __restrict__ dn0, const float* __restrict__ dn1,
                float* __restrict__ ag0, float* __restrict__ ag1) {
  const int e = blockIdx.y;
  const int lane = threadIdx.x & 31;
  const int edge = blockIdx.x * 8 + (threadIdx.x >> 5);
  const int* ed = e ? e1 : e0;
  const float* kqvS = e ? kqv1 : kqv0;
  const float* al = e ? al1 : al0;
  const float* dn = e ? dn0 : dn1;
  float* ag = e ? ag0 : ag1;   // dst type == 1-e
  int src = ed[edge], dst = ed[NEDGE + edge];
  int h = lane >> 3;
  float w = al[(size_t)edge * HEADS + h] / (dn[(size_t)dst * HEADS + h] + 1e-16f);
  int c = lane * 4;
  const float4 v4 = *(const float4*)(kqvS + (size_t)src * 3 * CH + 2 * CH + c);
  float* base = ag + (size_t)dst * CH + c;
  atomicAdd(base + 0, w * v4.x);
  atomicAdd(base + 1, w * v4.y);
  atomicAdd(base + 2, w * v4.z);
  atomicAdd(base + 3, w * v4.w);
}

// Fused: GELU(agg) @ W + b -> sigmoid-gated skip -> LayerNorm -> ReLU.
// One wave per 16-row tile; 8 accumulators cover all 128 output cols.
__global__ __launch_bounds__(32)
void outproj_kernel(const float* __restrict__ AGGt, const __bf16* __restrict__ Wp,
                    const float* __restrict__ bias, const float* __restrict__ xcur,
                    const float* __restrict__ skipp, const float* __restrict__ lnw,
                    const float* __restrict__ lnb, float* __restrict__ xnext) {
  __shared__ float ybuf[16 * CH];
  __shared__ float mu_s[16], rs_s[16];
  const int lane = threadIdx.x & 31;
  const int row0 = blockIdx.x * 16;

  f32x8 acc[8];
#pragma unroll
  for (int j = 0; j < 8; ++j) acc[j] = (f32x8){};

  for (int k0 = 0; k0 < CH; k0 += 32) {
    bf16x16 a = load_frag_a_gelu(AGGt, row0, CH, k0, lane);
#pragma unroll
    for (int j = 0; j < 8; ++j) {
      bf16x16 b = load_frag_b_packed(Wp, CH, j * 16, k0, lane);
      acc[j] = wmma_bf16(a, b, acc[j]);
    }
  }

  float sk = 1.0f / (1.0f + __expf(-skipp[0]));
  int n = lane & 15;
  int mb = (lane >> 4) << 3;
#pragma unroll
  for (int j = 0; j < 8; ++j) {
    float bv = bias[j * 16 + n];
#pragma unroll
    for (int r = 0; r < 8; ++r) {
      int row = mb + r, col = j * 16 + n;
      float y = acc[j][r] + bv;
      y = sk * y + (1.0f - sk) * xcur[(size_t)(row0 + row) * CH + col];
      ybuf[row * CH + col] = y;
    }
  }
  __syncthreads();
  if (lane < 16) {
    const float* r = &ybuf[lane * CH];
    float s = 0.0f, s2 = 0.0f;
    for (int c = 0; c < CH; ++c) { float v = r[c]; s += v; s2 += v * v; }
    float mu = s * (1.0f / CH);
    float var = s2 * (1.0f / CH) - mu * mu;
    mu_s[lane] = mu;
    rs_s[lane] = rsqrtf(var + 1e-5f);
  }
  __syncthreads();
  for (int i = lane; i < 16 * CH; i += 32) {
    int row = i >> 7, col = i & (CH - 1);
    float v = (ybuf[i] - mu_s[row]) * rs_s[row] * lnw[col] + lnb[col];
    xnext[(size_t)(row0 + row) * CH + col] = fmaxf(v, 0.0f);
  }
}

// ---------------- host launch ----------------
extern "C" void kernel_launch(void* const* d_in, const int* in_sizes, int n_in,
                              void* d_out, int out_size, void* d_ws, size_t ws_size,
                              hipStream_t stream) {
  (void)in_sizes; (void)n_in; (void)out_size; (void)ws_size;

  const float* x_user = (const float*)d_in[0];
  const float* x_item = (const float*)d_in[1];
  const int*   edge_ui = (const int*)d_in[2];
  const int*   edge_iu = (const int*)d_in[3];
  const float* kqv_w = (const float*)d_in[4];
  const float* kqv_b = (const float*)d_in[5];
  const float* out_w = (const float*)d_in[6];
  const float* out_b = (const float*)d_in[7];
  const float* a_rel = (const float*)d_in[8];
  const float* m_rel = (const float*)d_in[9];
  const float* p_rel = (const float*)d_in[10];
  const float* skip  = (const float*)d_in[11];
  const float* ln_w  = (const float*)d_in[12];
  const float* ln_b  = (const float*)d_in[13];
  float* out = (float*)d_out;

  float* ws = (float*)d_ws;
  const size_t NX = (size_t)NNODE * CH;        // 6.4M floats
  const size_t NKQV = (size_t)NNODE * 3 * CH;  // 19.2M floats
  float* XA0  = ws;
  float* XA1  = XA0 + NX;
  float* KQV0 = XA1 + NX;
  float* KQV1 = KQV0 + NKQV;
  float* AGG0 = KQV1 + NKQV;
  float* AGG1 = AGG0 + NX;
  float* AL0  = AGG1 + NX;
  float* AL1  = AL0 + (size_t)NEDGE * HEADS;
  float* AMAX0 = AL1 + (size_t)NEDGE * HEADS;
  float* AMAX1 = AMAX0 + (size_t)NNODE * HEADS;
  float* DEN0  = AMAX1 + (size_t)NNODE * HEADS;
  float* DEN1  = DEN0 + (size_t)NNODE * HEADS;
  // bf16 packed-weight regions (32B-aligned: 66.4M floats from 256B-aligned base)
  __bf16* PKQV = (__bf16*)(DEN1 + (size_t)NNODE * HEADS);     // 4 * 128*384
  __bf16* POUT = PKQV + (size_t)LAYERS * TYPES * CH * 3 * CH; // 4 * 128*128
  __bf16* PA   = POUT + (size_t)LAYERS * TYPES * CH * CH;     // 16 * 32*32
  __bf16* PM   = PA + (size_t)LAYERS * 2 * HEADS * HD * HD;   // 16 * 32*32

  // one-time weight packing (fp32 -> bf16 WMMA fragment order)
  pack_b_kernel<<<dim3(12, 4), 256, 0, stream>>>(kqv_w, PKQV, CH, 3 * CH);
  pack_b_kernel<<<dim3(4, 4), 256, 0, stream>>>(out_w, POUT, CH, CH);
  pack_b_kernel<<<dim3(1, 16), 256, 0, stream>>>(a_rel, PA, HD, HD);
  pack_b_kernel<<<dim3(1, 16), 256, 0, stream>>>(m_rel, PM, HD, HD);

  for (int l = 0; l < LAYERS; ++l) {
    const float* xc0 = l ? XA0 : x_user;
    const float* xc1 = l ? XA1 : x_item;
    float* xn0 = l ? out : XA0;
    float* xn1 = l ? (out + NX) : XA1;

    const int nAM = NNODE * HEADS * 2;
    fill_kernel<<<dim3((nAM + 255) / 256), 256, 0, stream>>>(AMAX0, -__builtin_inff(), nAM);
    fill_kernel<<<dim3((nAM + 255) / 256), 256, 0, stream>>>(DEN0, 0.0f, nAM);
    const int nAG = (int)(2 * NX);
    fill_kernel<<<dim3((nAG + 255) / 256), 256, 0, stream>>>(AGG0, 0.0f, nAG);

    gemm_bias_kernel<<<dim3(MT, 6), 128, 0, stream>>>(
        xc0, PKQV + (size_t)(l * 2 + 0) * CH * 3 * CH,
        kqv_b + (size_t)(l * 2 + 0) * 3 * CH, KQV0, CH, CH, 3 * CH);
    gemm_bias_kernel<<<dim3(MT, 6), 128, 0, stream>>>(
        xc1, PKQV + (size_t)(l * 2 + 1) * CH * 3 * CH,
        kqv_b + (size_t)(l * 2 + 1) * 3 * CH, KQV1, CH, CH, 3 * CH);

    rel_kernel<<<dim3(MT, HEADS, 4), 32, 0, stream>>>(KQV0, KQV1, PA, PM, l);

    alpha_kernel<<<dim3(NEDGE / 8, 2), 256, 0, stream>>>(
        KQV0, KQV1, edge_ui, edge_iu, p_rel, AL0, AL1, AMAX0, AMAX1, l);
    exp_kernel<<<dim3(NEDGE * HEADS / 256, 2), 256, 0, stream>>>(
        edge_ui, edge_iu, AL0, AL1, AMAX0, AMAX1, DEN0, DEN1);
    agg_kernel<<<dim3(NEDGE / 8, 2), 256, 0, stream>>>(
        KQV0, KQV1, edge_ui, edge_iu, AL0, AL1, DEN0, DEN1, AGG0, AGG1);

    outproj_kernel<<<dim3(MT), 32, 0, stream>>>(
        AGG0, POUT + (size_t)(l * 2 + 0) * CH * CH, out_b + (size_t)(l * 2 + 0) * CH,
        xc0, skip + (l * 2 + 0), ln_w + (size_t)(l * 2 + 0) * CH,
        ln_b + (size_t)(l * 2 + 0) * CH, xn0);
    outproj_kernel<<<dim3(MT), 32, 0, stream>>>(
        AGG1, POUT + (size_t)(l * 2 + 1) * CH * CH, out_b + (size_t)(l * 2 + 1) * CH,
        xc1, skip + (l * 2 + 1), ln_w + (size_t)(l * 2 + 1) * CH,
        ln_b + (size_t)(l * 2 + 1) * CH, xn1);
  }
}
